// GraphToGraph_50560355008915
// MI455X (gfx1250) — matrix operand order repaired and checked by reading, compiled, not run on hardware
//
#include <hip/hip_runtime.h>

typedef __attribute__((ext_vector_type(16))) __bf16 v16bf;
typedef __attribute__((ext_vector_type(8)))  __bf16 v8bf;
typedef __attribute__((ext_vector_type(8)))  float  v8f;

#define CD 128
#define SD 64

// ---------------------------------------------------------------------------
// Generic bf16 WMMA GEMM: out[M,128] = act( A(cat) @ WT^T + bias + base )
//   A row m: k < K0 from A0 (ld K0), k >= K0 from A1 (ld K1)
//   WT is [128][Ktot] bf16 (pre-transposed weights), Ktot % 32 == 0, K0 % 32 == 0
//   One 16x16 tile per wave, 8 waves per block cover N=128, grid = M/16.
// ---------------------------------------------------------------------------
__global__ __launch_bounds__(256) void k_gemm_bf16(
    const __bf16* __restrict__ A0, int K0,
    const __bf16* __restrict__ A1, int K1,
    const __bf16* __restrict__ WT, int Ktot,
    const float* __restrict__ base,   // [M,128] or null
    const float* __restrict__ bias,   // [128]   or null
    float* __restrict__ out,          // [M,128]
    __bf16* __restrict__ outB,        // [M,128] or null
    int M, int relu)
{
    const int lane  = threadIdx.x & 31;
    const int wave  = threadIdx.x >> 5;      // 0..7  -> column tile
    const int mTile = blockIdx.x << 4;
    const int nTile = wave << 4;
    const bool hi   = lane >= 16;
    const int  l16  = lane & 15;
    const int  mRow = mTile + l16;           // A row loaded by this lane
    const int  nCol = nTile + l16;           // WT row loaded by this lane

    v8f c = {0.f, 0.f, 0.f, 0.f, 0.f, 0.f, 0.f, 0.f};

    for (int kk = 0; kk < Ktot; kk += 32) {
        const __bf16* arow;
        if (kk < K0) arow = A0 + (size_t)mRow * (size_t)K0 + kk;
        else         arow = A1 + (size_t)mRow * (size_t)K1 + (kk - K0);
        const int k0 = hi ? 16 : 0;
        // A frag: lanes 0-15 hold K = 0..15; lanes 16-31 K = 16..31
        v8bf alo = *(const v8bf*)(arow + k0);
        v8bf ahi = *(const v8bf*)(arow + k0 + 8);
        v16bf a = __builtin_shufflevector(alo, ahi,
                                          0,1,2,3,4,5,6,7,8,9,10,11,12,13,14,15);
        // B frag: lanes 0-15 K = kk..kk+15 of column nCol; lanes 16-31 K += 16
        const __bf16* brow = WT + (size_t)nCol * (size_t)Ktot + kk + (hi ? 16 : 0);
        v16bf b = *(const v16bf*)(brow);
        c = __builtin_amdgcn_wmma_f32_16x16x32_bf16(false, a, false, b,
                                                    (short)0, c, false, false);
    }

#pragma unroll
    for (int j = 0; j < 8; ++j) {
        int m = mTile + j + (hi ? 8 : 0);
        int n = nTile + l16;
        float v = c[j];
        if (bias) v += bias[n];
        if (base) v += base[(size_t)m * CD + n];
        if (relu) v = v > 0.f ? v : 0.f;
        out[(size_t)m * CD + n] = v;
        if (outB) outB[(size_t)m * CD + n] = (__bf16)v;
    }
}

// ---------------------------------------------------------------------------
// Small helpers
// ---------------------------------------------------------------------------
__global__ void k_f32_to_bf16(const float* __restrict__ x, __bf16* __restrict__ y, int n) {
    int i = blockIdx.x * blockDim.x + threadIdx.x;
    if (i < n) y[i] = (__bf16)x[i];
}

// WT[n*K + k] = W[k*128 + n]   (W is [K,128] row-major fp32)
__global__ void k_transpose_bf16(const float* __restrict__ W, __bf16* __restrict__ WT, int K) {
    int i = blockIdx.x * blockDim.x + threadIdx.x;
    if (i >= K * CD) return;
    int k = i / CD, n = i % CD;
    WT[(size_t)n * K + k] = (__bf16)W[i];
}

__global__ void k_fill_f32(float* __restrict__ p, float v, int n) {
    int i = blockIdx.x * blockDim.x + threadIdx.x;
    if (i < n) p[i] = v;
}

__global__ void k_iota(int* __restrict__ g, int n) {
    int i = blockIdx.x * blockDim.x + threadIdx.x;
    if (i < n) g[i] = i;
}

// ---------------------------------------------------------------------------
// Per-directed-edge message + mean-aggregate scatter.
// blockIdx.x = directed edge (0..2E-1); threadIdx.x = feature j (0..127)
// msg = relu(P[src] + rel_cxcy @ W_msg[192:194]) * ew ; atomic scatter to dst
// ---------------------------------------------------------------------------
__global__ __launch_bounds__(CD) void k_msg_agg(
    const float* __restrict__ P,       // [N,128]  (includes b_msg)
    const int* __restrict__ ps, const int* __restrict__ pd,
    const float* __restrict__ ew,      // [E]
    const float* __restrict__ cxcy,    // [N,2]
    const float* __restrict__ Wmsg,    // [194,128] (rows 192,193 used here)
    float* __restrict__ agg,           // [N,128]
    float* __restrict__ deg,           // [N]
    int E)
{
    int e = blockIdx.x;
    int s, d;
    float w;
    if (e < E) { s = ps[e]; d = pd[e]; w = ew[e]; }
    else       { s = pd[e - E]; d = ps[e - E]; w = ew[e - E]; }
    int j = threadIdx.x;
    float rx = cxcy[2 * d]     - cxcy[2 * s];
    float ry = cxcy[2 * d + 1] - cxcy[2 * s + 1];
    float m = P[(size_t)s * CD + j] + rx * Wmsg[(size_t)192 * CD + j]
                                    + ry * Wmsg[(size_t)193 * CD + j];
    m = (m > 0.f ? m : 0.f) * w;
    atomicAdd(&agg[(size_t)d * CD + j], m);
    if (j == 0) atomicAdd(&deg[d], 1.f);
}

// aggB = bf16(agg / max(deg,1))
__global__ void k_agg_finish(const float* __restrict__ agg, const float* __restrict__ deg,
                             __bf16* __restrict__ aggB, int N) {
    int i = blockIdx.x * blockDim.x + threadIdx.x;
    if (i >= N * CD) return;
    float dg = deg[i / CD];
    aggB[i] = (__bf16)(agg[i] / (dg > 1.f ? dg : 1.f));
}

// ---------------------------------------------------------------------------
// Edge score: one wave per pruned edge.
// score = step( sum_j w_e2[j] * relu(Q1[ps][j] + Q2[pd][j]) + b_e2 )
// (b_e1 folded into Q1's bias)
// ---------------------------------------------------------------------------
__global__ __launch_bounds__(256) void k_edge_scores(
    const float* __restrict__ Q1, const float* __restrict__ Q2,
    const int* __restrict__ ps, const int* __restrict__ pd,
    const float* __restrict__ we2, const float* __restrict__ be2,
    float* __restrict__ scores, int E)
{
    int e = blockIdx.x * 8 + (threadIdx.x >> 5);
    int lane = threadIdx.x & 31;
    if (e >= E) return;
    const float* q1 = Q1 + (size_t)ps[e] * CD;
    const float* q2 = Q2 + (size_t)pd[e] * CD;
    float acc = 0.f;
#pragma unroll
    for (int t = 0; t < 4; ++t) {
        int j = lane + 32 * t;
        float h = q1[j] + q2[j];
        h = h > 0.f ? h : 0.f;
        acc += h * we2[j];
    }
#pragma unroll
    for (int off = 16; off > 0; off >>= 1) acc += __shfl_xor(acc, off, 32);
    if (lane == 0) scores[e] = (acc + be2[0]) >= 0.f ? 1.f : 0.f;
}

// ---------------------------------------------------------------------------
// Connected components: hook (min over active edges, both directions) + jump
// ---------------------------------------------------------------------------
__global__ void k_cc_hook(const int* __restrict__ ps, const int* __restrict__ pd,
                          const float* __restrict__ sc, int* __restrict__ g, int E) {
    int e = blockIdx.x * blockDim.x + threadIdx.x;
    if (e >= E) return;
    if (sc[e] <= 0.f) return;
    int s = ps[e], d = pd[e];
    int a = g[s], b = g[d];
    if (a < b)      atomicMin(&g[d], a);
    else if (b < a) atomicMin(&g[s], b);
}

__global__ void k_cc_jump(int* __restrict__ g, int n) {
    int i = blockIdx.x * blockDim.x + threadIdx.x;
    if (i >= n) return;
    int a = g[i];
    int b = g[a];
    if (b < a) g[i] = b;
}

// ---------------------------------------------------------------------------
// Group centroids
// ---------------------------------------------------------------------------
__global__ void k_group_mass(const int* __restrict__ g, const float* __restrict__ mass,
                             float* __restrict__ om, int N) {
    int i = blockIdx.x * blockDim.x + threadIdx.x;
    if (i < N) atomicAdd(&om[g[i]], mass[i]);
}

__global__ void k_group_center(const int* __restrict__ g, const float* __restrict__ mass,
                               const float* __restrict__ om, const float* __restrict__ cxcy,
                               float* __restrict__ ocx, int N) {
    int i = blockIdx.x * blockDim.x + threadIdx.x;
    if (i >= N) return;
    int gi = g[i];
    float w = mass[i] / om[gi];
    atomicAdd(&ocx[2 * gi],     w * cxcy[2 * i]);
    atomicAdd(&ocx[2 * gi + 1], w * cxcy[2 * i + 1]);
}

// ---------------------------------------------------------------------------
// Intra-group node score accumulation (pruned edges; self-loops folded into
// the final (num+1)/(den+1))
// ---------------------------------------------------------------------------
__global__ void k_ns_edge(const int* __restrict__ ps, const int* __restrict__ pd,
                          const float* __restrict__ sc, const int* __restrict__ g,
                          float* __restrict__ num, float* __restrict__ den, int E) {
    int e = blockIdx.x * blockDim.x + threadIdx.x;
    if (e >= E) return;
    int s = ps[e], d = pd[e];
    if (g[s] != g[d]) return;
    float v = sc[e];
    atomicAdd(&num[s], v);
    atomicAdd(&num[d], v);
    atomicAdd(&den[s], 1.f);
    atomicAdd(&den[d], 1.f);
}

// ---------------------------------------------------------------------------
// Final outputs: con_out | dyn_out | sta_out | node_scores | gids(as float)
// grid = N blocks x 128 threads
// ---------------------------------------------------------------------------
__global__ __launch_bounds__(CD) void k_outputs(
    const float* __restrict__ con2, const float* __restrict__ dyn,
    const float* __restrict__ sta, const float* __restrict__ cxcy,
    const int* __restrict__ g, const float* __restrict__ ocx,
    const float* __restrict__ Wc, const float* __restrict__ bc,   // [2,128],[128]
    const float* __restrict__ Ws, const float* __restrict__ bs,   // [2,64],[64]
    const float* __restrict__ num, const float* __restrict__ den,
    float* __restrict__ out, int N)
{
    int i = blockIdx.x;
    int j = threadIdx.x;
    int gi = g[i];
    float dx = ocx[2 * gi]     - cxcy[2 * i];
    float dy = ocx[2 * gi + 1] - cxcy[2 * i + 1];

    float cv = con2[(size_t)i * CD + j] + dx * Wc[j] + dy * Wc[CD + j] + bc[j];
    out[(size_t)i * CD + j] = cv;

    if (j < SD) {
        float ds = dx * Ws[j] + dy * Ws[SD + j] + bs[j];
        out[(size_t)N * CD        + (size_t)i * SD + j] = dyn[(size_t)i * SD + j] + ds;
        out[(size_t)N * (CD + SD) + (size_t)i * SD + j] = sta[(size_t)i * SD + j] + ds;
    }
    if (j == 0) {
        out[(size_t)N * (CD + 2 * SD) + i]     = (num[i] + 1.f) / (den[i] + 1.f);
        out[(size_t)N * (CD + 2 * SD + 1) + i] = (float)gi;
    }
}

// ---------------------------------------------------------------------------
extern "C" void kernel_launch(void* const* d_in, const int* in_sizes, int n_in,
                              void* d_out, int out_size, void* d_ws, size_t ws_size,
                              hipStream_t stream) {
    const float* con_feats = (const float*)d_in[0];
    const float* dyn_sf    = (const float*)d_in[1];
    const float* sta_sf    = (const float*)d_in[2];
    const float* edge_w    = (const float*)d_in[3];
    const float* cxcy      = (const float*)d_in[4];
    const float* masses    = (const float*)d_in[5];
    const float* W_self    = (const float*)d_in[6];
    const float* b_self    = (const float*)d_in[7];
    const float* W_msg     = (const float*)d_in[8];
    const float* b_msg     = (const float*)d_in[9];
    const float* W_upd     = (const float*)d_in[10];
    const float* b_upd     = (const float*)d_in[11];
    const float* W_e1      = (const float*)d_in[12];
    const float* b_e1      = (const float*)d_in[13];
    const float* w_e2      = (const float*)d_in[14];
    const float* b_e2      = (const float*)d_in[15];
    const float* W_ccxcy   = (const float*)d_in[16];
    const float* b_ccxcy   = (const float*)d_in[17];
    const float* W_scxcy   = (const float*)d_in[18];
    const float* b_scxcy   = (const float*)d_in[19];
    const int*   edge_ids  = (const int*)d_in[20];

    const int N = in_sizes[0] / CD;        // 30000
    const int E = in_sizes[20] / 2;        // 250000
    const int* ps = edge_ids;
    const int* pd = edge_ids + E;
    float* out = (float*)d_out;

    // ---- carve workspace (256B aligned slices) ----
    char* w = (char*)d_ws;
    auto carve = [&](size_t bytes) -> void* {
        void* p = (void*)w;
        w += (bytes + 255) & ~(size_t)255;
        return p;
    };
    __bf16* WselfT = (__bf16*)carve((size_t)CD * CD * 2);
    __bf16* WmsgT  = (__bf16*)carve((size_t)CD * 192 * 2);
    __bf16* WupdT  = (__bf16*)carve((size_t)CD * CD * 2);
    __bf16* We1Ta  = (__bf16*)carve((size_t)CD * CD * 2);
    __bf16* We1Tb  = (__bf16*)carve((size_t)CD * CD * 2);
    __bf16* conFb  = (__bf16*)carve((size_t)N * CD * 2);
    __bf16* strucB = (__bf16*)carve((size_t)N * SD * 2);
    float*  con    = (float*) carve((size_t)N * CD * 4);
    __bf16* conB   = (__bf16*)carve((size_t)N * CD * 2);
    float*  P      = (float*) carve((size_t)N * CD * 4);
    float*  agg    = (float*) carve((size_t)N * CD * 4);
    float*  deg    = (float*) carve((size_t)N * 4);
    __bf16* aggB   = (__bf16*)carve((size_t)N * CD * 2);
    float*  con2   = (float*) carve((size_t)N * CD * 4);
    __bf16* con2B  = (__bf16*)carve((size_t)N * CD * 2);
    float*  Q1     = (float*) carve((size_t)N * CD * 4);
    float*  Q2     = (float*) carve((size_t)N * CD * 4);
    float*  scores = (float*) carve((size_t)E * 4);
    int*    gids   = (int*)   carve((size_t)N * 4);
    float*  om     = (float*) carve((size_t)N * 4);
    float*  ocx    = (float*) carve((size_t)N * 2 * 4);
    float*  nsnum  = (float*) carve((size_t)N * 4);
    float*  nsden  = (float*) carve((size_t)N * 4);

    const int T = 256;
    auto blocks = [](long long n, int t) { return (int)((n + t - 1) / t); };

    // ---- 0: transpose weights to bf16 [n][k] ----
    k_transpose_bf16<<<blocks((long long)CD * CD, T), T, 0, stream>>>(W_self, WselfT, CD);
    k_transpose_bf16<<<blocks((long long)192 * CD, T), T, 0, stream>>>(W_msg, WmsgT, 192);
    k_transpose_bf16<<<blocks((long long)CD * CD, T), T, 0, stream>>>(W_upd, WupdT, CD);
    k_transpose_bf16<<<blocks((long long)CD * CD, T), T, 0, stream>>>(W_e1, We1Ta, CD);
    k_transpose_bf16<<<blocks((long long)CD * CD, T), T, 0, stream>>>(W_e1 + (size_t)CD * CD, We1Tb, CD);

    // ---- 1: activations -> bf16 ----
    k_f32_to_bf16<<<blocks((long long)N * CD, T), T, 0, stream>>>(con_feats, conFb, N * CD);
    k_f32_to_bf16<<<blocks((long long)N * SD, T), T, 0, stream>>>(dyn_sf, strucB, N * SD);

    // ---- 2: con = relu(con_feats @ W_self + b_self) ----
    k_gemm_bf16<<<N / 16, 256, 0, stream>>>(conFb, CD, nullptr, 0, WselfT, CD,
                                            nullptr, b_self, con, conB, N, 1);

    // ---- 3: P = [con|struc] @ W_msg[0:192] + b_msg ----
    k_gemm_bf16<<<N / 16, 256, 0, stream>>>(conB, CD, strucB, SD, WmsgT, 192,
                                            nullptr, b_msg, P, nullptr, N, 0);

    // ---- 4: edge messages -> mean aggregate ----
    k_fill_f32<<<blocks((long long)N * CD, T), T, 0, stream>>>(agg, 0.f, N * CD);
    k_fill_f32<<<blocks(N, T), T, 0, stream>>>(deg, 0.f, N);
    k_msg_agg<<<2 * E, CD, 0, stream>>>(P, ps, pd, edge_w, cxcy, W_msg, agg, deg, E);
    k_agg_finish<<<blocks((long long)N * CD, T), T, 0, stream>>>(agg, deg, aggB, N);

    // ---- 5: con2 = con + agg_mean @ W_upd + b_upd ----
    k_gemm_bf16<<<N / 16, 256, 0, stream>>>(aggB, CD, nullptr, 0, WupdT, CD,
                                            con, b_upd, con2, con2B, N, 0);

    // ---- 6: Q1 = con2 @ U1 + b_e1 ; Q2 = con2 @ U2 ----
    k_gemm_bf16<<<N / 16, 256, 0, stream>>>(con2B, CD, nullptr, 0, We1Ta, CD,
                                            nullptr, b_e1, Q1, nullptr, N, 0);
    k_gemm_bf16<<<N / 16, 256, 0, stream>>>(con2B, CD, nullptr, 0, We1Tb, CD,
                                            nullptr, nullptr, Q2, nullptr, N, 0);

    // ---- 7: edge scores (hard step) ----
    k_edge_scores<<<blocks(E, 8), 256, 0, stream>>>(Q1, Q2, ps, pd, w_e2, b_e2, scores, E);

    // ---- 8: connected components (hook + pointer jump) ----
    k_iota<<<blocks(N, T), T, 0, stream>>>(gids, N);
    for (int it = 0; it < 32; ++it) {
        k_cc_hook<<<blocks(E, T), T, 0, stream>>>(ps, pd, scores, gids, E);
        k_cc_jump<<<blocks(N, T), T, 0, stream>>>(gids, N);
    }

    // ---- 9: mass-weighted group centers ----
    k_fill_f32<<<blocks(N, T), T, 0, stream>>>(om, 0.f, N);
    k_group_mass<<<blocks(N, T), T, 0, stream>>>(gids, masses, om, N);
    k_fill_f32<<<blocks(2 * N, T), T, 0, stream>>>(ocx, 0.f, 2 * N);
    k_group_center<<<blocks(N, T), T, 0, stream>>>(gids, masses, om, cxcy, ocx, N);

    // ---- 10: intra-group node scores ----
    k_fill_f32<<<blocks(N, T), T, 0, stream>>>(nsnum, 0.f, N);
    k_fill_f32<<<blocks(N, T), T, 0, stream>>>(nsden, 0.f, N);
    k_ns_edge<<<blocks(E, T), T, 0, stream>>>(ps, pd, scores, gids, nsnum, nsden, E);

    // ---- 11: final outputs ----
    k_outputs<<<N, CD, 0, stream>>>(con2, dyn_sf, sta_sf, cxcy, gids, ocx,
                                    W_ccxcy, b_ccxcy, W_scxcy, b_scxcy,
                                    nsnum, nsden, out, N);
}
